// DLRM_Net_59682865545859
// MI455X (gfx1250) — matrix operand order repaired and checked by reading, not compiled, not run on hardware
//
#include <hip/hip_runtime.h>
#include <hip/hip_bf16.h>
#include <math.h>

// ---------------------------------------------------------------------------
// DLRM forward for MI455X (gfx1250): f16 WMMA GEMMs with f32 accumulate.
// ---------------------------------------------------------------------------

typedef __attribute__((ext_vector_type(16))) _Float16 v16h;
typedef __attribute__((ext_vector_type(8)))  _Float16 v8h;
typedef __attribute__((ext_vector_type(4)))  _Float16 v4h;
typedef __attribute__((ext_vector_type(8)))  float    v8f;

#define BATCH      16384
#define NTAB       26
#define VOCAB      100000
#define EMB_D      64
#define T_STRIDE   (27 * 64)    // 1728 halfs per sample
#define R_STRIDE   416          // 415 padded to 13 K-chunks of 32

__device__ __forceinline__ int imin(int a, int b) { return a < b ? a : b; }

// Concatenate two contiguous 8-half loads into one A/B fragment (pure SSA;
// avoids any compiler demotion of fragment arrays to memory).
__device__ __forceinline__ v16h cat16(v8h lo, v8h hi) {
    return __builtin_shufflevector(lo, hi, 0, 1, 2, 3, 4, 5, 6, 7,
                                           8, 9, 10, 11, 12, 13, 14, 15);
}
// A fragment (16x32, lane L = row L%16): halfs [k0*32+16*hl .. +15]
__device__ __forceinline__ v16h load_afrag(const _Float16* p) {
    return cat16(*(const v8h*)(p), *(const v8h*)(p + 8));
}

// ---------------------------------------------------------------------------
// Pack f32 weight W (Mout x Kact, row-major) into per-lane WMMA-B layout,
// K zero-padded to Kc*32.  B frag, lane L, element e:
//   K = e + 16*(L>=16),  N = L%16   (contiguous k order used by this code, matching load_afrag)
// Linear packed index: (((n0*Kc + k0)*32 + lane)*16 + e)
// ---------------------------------------------------------------------------
__global__ __launch_bounds__(256)
void pack_weight(const float* __restrict__ W, int Kact, int Kc,
                 _Float16* __restrict__ Wp, int total)
{
    int tid = blockIdx.x * blockDim.x + threadIdx.x;
    if (tid >= total) return;
    int e    = tid & 15;
    int lane = (tid >> 4) & 31;
    int blk  = tid >> 9;
    int k0   = blk % Kc;
    int n0   = blk / Kc;
    int hl   = lane >> 4;
    int kk   = k0 * 32 + e + 16 * hl;
    int nn   = n0 * 16 + (lane & 15);
    float v  = (kk < Kact) ? W[(size_t)nn * Kact + kk] : 0.0f;
    Wp[tid]  = (_Float16)v;
}

// dense_x (B x 13, f32) -> A0 (B x 32, f16, zero padded)
__global__ __launch_bounds__(256)
void cvt_dense(const float* __restrict__ X, _Float16* __restrict__ A0)
{
    int tid = blockIdx.x * blockDim.x + threadIdx.x;   // B*32 threads
    int b = tid >> 5, k = tid & 31;
    float v = (k < 13) ? X[b * 13 + k] : 0.0f;
    A0[tid] = (_Float16)v;
}

// ---------------------------------------------------------------------------
// WMMA GEMM:  C(f16, ReLU) = A(f16, rows x KC*32) * Wp + bias
// Block = 128 threads (4 waves); each wave -> 32 rows x 64 cols
// (2 A-frags x 4 B-frags -> 8 WMMAs per K=32 step).
// grid.x = rows/128, grid.y = N/64.   KC is compile-time: the K loop fully
// unrolls, all fragments stay in registers, and the scheduler pipelines the
// next step's global loads under the current step's WMMAs.
// ---------------------------------------------------------------------------
template <int KC>
__global__ __launch_bounds__(128)
void gemm_wmma_relu(const _Float16* __restrict__ A, int lda,
                    const _Float16* __restrict__ Wp,
                    const float* __restrict__ bias,
                    _Float16* __restrict__ C, int ldc)
{
    const int lane   = threadIdx.x & 31;
    const int wave   = threadIdx.x >> 5;
    const int hl     = lane >> 4;
    const int lane16 = lane & 15;
    const int row0   = (blockIdx.x * 4 + wave) * 32;   // 32 rows per wave
    const int nt0    = blockIdx.y * 4;                 // 4 col tiles per wave

    v8f acc[2][4];
#pragma unroll
    for (int mi = 0; mi < 2; ++mi)
#pragma unroll
        for (int t = 0; t < 4; ++t)
#pragma unroll
            for (int r = 0; r < 8; ++r) acc[mi][t][r] = 0.0f;

    const _Float16* arow0 = A + (size_t)(row0 + lane16)      * lda + hl * 16;
    const _Float16* arow1 = A + (size_t)(row0 + 16 + lane16) * lda + hl * 16;
    const _Float16* wbase = Wp + (((size_t)nt0 * KC * 32 + lane) << 4);
    const size_t wstep = (size_t)KC * 512;   // halfs between adjacent n-tiles

#pragma unroll
    for (int k0 = 0; k0 < KC; ++k0) {
        const v16h a0 = load_afrag(arow0 + k0 * 32);
        const v16h a1 = load_afrag(arow1 + k0 * 32);
        const _Float16* wp = wbase + ((size_t)k0 << 9);   // + k0 * 32 lanes * 16
#pragma unroll
        for (int t = 0; t < 4; ++t) {
            const v16h bf = *(const v16h*)(wp + (size_t)t * wstep);
            acc[0][t] = __builtin_amdgcn_wmma_f32_16x16x32_f16(
                false, a0, false, bf, (short)0, acc[0][t], false, false);
            acc[1][t] = __builtin_amdgcn_wmma_f32_16x16x32_f16(
                false, a1, false, bf, (short)0, acc[1][t], false, false);
        }
    }

#pragma unroll
    for (int t = 0; t < 4; ++t) {
        const int col = (nt0 + t) * 16 + lane16;
        const float bb = bias[col];
#pragma unroll
        for (int mi = 0; mi < 2; ++mi) {
#pragma unroll
            for (int r = 0; r < 8; ++r) {
                const int row = row0 + mi * 16 + r + 8 * hl;
                float v = acc[mi][t][r] + bb;
                v = v > 0.0f ? v : 0.0f;
                C[(size_t)row * ldc + col] = (_Float16)v;
            }
        }
    }
}

// ---------------------------------------------------------------------------
// Embedding gather: T[b][1+t][:] = (f16) emb[t][lS_i[t][b]][:]
// One thread handles 4 consecutive elements (float4 in, v4h out).
// ---------------------------------------------------------------------------
__global__ __launch_bounds__(256)
void gather_emb(const int* __restrict__ lS_i, const float* __restrict__ emb,
                _Float16* __restrict__ T)
{
    int tid = blockIdx.x * blockDim.x + threadIdx.x;   // NTAB*B*16 threads
    int q  = tid & 15;
    int bt = tid >> 4;
    int b  = bt & (BATCH - 1);
    int t  = bt >> 14;                                 // B = 2^14
    if (t >= NTAB) return;
    int idx = lS_i[t * BATCH + b];
    const float4 s = *(const float4*)(emb +
        ((size_t)t * VOCAB + (size_t)idx) * EMB_D + q * 4);
    v4h d;
    d[0] = (_Float16)s.x; d[1] = (_Float16)s.y;
    d[2] = (_Float16)s.z; d[3] = (_Float16)s.w;
    *(v4h*)(T + (size_t)b * T_STRIDE + (1 + t) * EMB_D + q * 4) = d;
}

// ---------------------------------------------------------------------------
// Interaction: per sample b, Z = T * T^T (27x27 over D=64), extract lower
// triangle (i>j) into R[b][64 + i*(i-1)/2 + j]; also copy x row into R[b][0:64]
// and zero pad column 415.  One wave per sample; 3 output tiles * 2 K-steps
// = 6 WMMAs.  Rows clamped to 26 (padding rows never extracted).
// ---------------------------------------------------------------------------
__global__ __launch_bounds__(256)
void interact_wmma(const _Float16* __restrict__ T, _Float16* __restrict__ R)
{
    const int lane   = threadIdx.x & 31;
    const int wave   = threadIdx.x >> 5;
    const int b      = blockIdx.x * 8 + wave;
    const int hl     = lane >> 4;
    const int lane16 = lane & 15;

    const _Float16* Tb = T + (size_t)b * T_STRIDE;
    _Float16*       Rb = R + (size_t)b * R_STRIDE;

    // R[b][0:64] = x = T[b][0][:]
    *(uint32_t*)(Rb + 2 * lane) = *(const uint32_t*)(Tb + 2 * lane);
    if (lane == 0) Rb[415] = (_Float16)0.0f;

    // A fragments: rows, clamped to 26 (padding rows never extracted)
    const _Float16* tpa0 = Tb + imin(lane16, 26)      * EMB_D + hl * 16;
    const _Float16* tpa1 = Tb + imin(16 + lane16, 26) * EMB_D + hl * 16;
    const v16h a00 = load_afrag(tpa0);
    const v16h a01 = load_afrag(tpa0 + 32);
    const v16h a10 = load_afrag(tpa1);
    const v16h a11 = load_afrag(tpa1 + 32);

    // B fragments: column n is contiguous over K in T row n.
    const _Float16* tpb0 = Tb + imin(lane16, 26)      * EMB_D + 16 * hl;
    const _Float16* tpb1 = Tb + imin(16 + lane16, 26) * EMB_D + 16 * hl;
    const v16h b00 = *(const v16h*)(tpb0);
    const v16h b01 = *(const v16h*)(tpb0 + 32);
    const v16h b10 = *(const v16h*)(tpb1);
    const v16h b11 = *(const v16h*)(tpb1 + 32);

    v8f acc00, acc10, acc11;
#pragma unroll
    for (int r = 0; r < 8; ++r) { acc00[r] = 0.f; acc10[r] = 0.f; acc11[r] = 0.f; }

    acc00 = __builtin_amdgcn_wmma_f32_16x16x32_f16(false, a00, false, b00, (short)0, acc00, false, false);
    acc00 = __builtin_amdgcn_wmma_f32_16x16x32_f16(false, a01, false, b01, (short)0, acc00, false, false);
    acc10 = __builtin_amdgcn_wmma_f32_16x16x32_f16(false, a10, false, b00, (short)0, acc10, false, false);
    acc10 = __builtin_amdgcn_wmma_f32_16x16x32_f16(false, a11, false, b01, (short)0, acc10, false, false);
    acc11 = __builtin_amdgcn_wmma_f32_16x16x32_f16(false, a10, false, b10, (short)0, acc11, false, false);
    acc11 = __builtin_amdgcn_wmma_f32_16x16x32_f16(false, a11, false, b11, (short)0, acc11, false, false);

    // Extract: D element r of lane L -> (M = r + 8*hl, N = L%16)
#pragma unroll
    for (int r = 0; r < 8; ++r) {
        int i, j;
        i = r + 8 * hl;        j = lane16;            // tile (0,0)
        if (i < 27 && j < i) Rb[64 + i * (i - 1) / 2 + j] = (_Float16)acc00[r];
        i = 16 + r + 8 * hl;   j = lane16;            // tile (1,0)
        if (i < 27 && j < i) Rb[64 + i * (i - 1) / 2 + j] = (_Float16)acc10[r];
        i = 16 + r + 8 * hl;   j = 16 + lane16;       // tile (1,1)
        if (i < 27 && j < i) Rb[64 + i * (i - 1) / 2 + j] = (_Float16)acc11[r];
    }
}

// Final layer: out[b] = sigmoid( dot(th2[b], w) + bias )
__global__ __launch_bounds__(256)
void final_sigmoid(const _Float16* __restrict__ X, const float* __restrict__ w,
                   const float* __restrict__ bias, float* __restrict__ out)
{
    int b = blockIdx.x * blockDim.x + threadIdx.x;
    const _Float16* x = X + (size_t)b * 256;
    float s = bias[0];
#pragma unroll 8
    for (int k = 0; k < 256; ++k) s += (float)x[k] * w[k];
    out[b] = 1.0f / (1.0f + expf(-s));
}

// ---------------------------------------------------------------------------
extern "C" void kernel_launch(void* const* d_in, const int* in_sizes, int n_in,
                              void* d_out, int out_size, void* d_ws, size_t ws_size,
                              hipStream_t stream)
{
    (void)in_sizes; (void)n_in; (void)out_size; (void)ws_size;

    const float* dense_x = (const float*)d_in[0];
    const int*   lS_i    = (const int*)  d_in[2];   // d_in[1] = lS_o (identity, unused)
    const float* emb     = (const float*)d_in[3];
    const float* bW0 = (const float*)d_in[4];   // 512 x 13
    const float* bW1 = (const float*)d_in[5];   // 256 x 512
    const float* bW2 = (const float*)d_in[6];   // 64  x 256
    const float* bb0 = (const float*)d_in[7];
    const float* bb1 = (const float*)d_in[8];
    const float* bb2 = (const float*)d_in[9];
    const float* tW0 = (const float*)d_in[10];  // 512 x 415
    const float* tW1 = (const float*)d_in[11];  // 256 x 512
    const float* tW2 = (const float*)d_in[12];  // 1   x 256
    const float* tb0 = (const float*)d_in[13];
    const float* tb1 = (const float*)d_in[14];
    const float* tb2 = (const float*)d_in[15];
    float* out = (float*)d_out;

    char* ws = (char*)d_ws;
    size_t off = 0;
    auto alloc = [&](size_t bytes) -> void* {
        off = (off + 255) & ~(size_t)255;
        void* p = ws + off;
        off += bytes;
        return p;
    };

    _Float16* Wb0p = (_Float16*)alloc((size_t)512 * 32  * 2);
    _Float16* Wb1p = (_Float16*)alloc((size_t)256 * 512 * 2);
    _Float16* Wb2p = (_Float16*)alloc((size_t)64  * 256 * 2);
    _Float16* Wt0p = (_Float16*)alloc((size_t)512 * 416 * 2);
    _Float16* Wt1p = (_Float16*)alloc((size_t)256 * 512 * 2);
    _Float16* A0   = (_Float16*)alloc((size_t)BATCH * 32  * 2);
    _Float16* h1   = (_Float16*)alloc((size_t)BATCH * 512 * 2);
    _Float16* h2   = (_Float16*)alloc((size_t)BATCH * 256 * 2);
    _Float16* T    = (_Float16*)alloc((size_t)BATCH * T_STRIDE * 2);
    _Float16* R    = (_Float16*)alloc((size_t)BATCH * R_STRIDE * 2);
    _Float16* th1  = (_Float16*)alloc((size_t)BATCH * 512 * 2);
    _Float16* th2  = (_Float16*)alloc((size_t)BATCH * 256 * 2);

    // ---- weight packing (f32 -> f16, WMMA-B fragment layout) ----
    auto packs = [&](const float* W, int Kact, int Kc, _Float16* Wp, int Mout) {
        const int total = (Mout / 16) * Kc * 512;   // one thread per packed half
        hipLaunchKernelGGL(pack_weight, dim3((total + 255) / 256), dim3(256), 0, stream,
                           W, Kact, Kc, Wp, total);
    };
    packs(bW0, 13,  1,  Wb0p, 512);
    packs(bW1, 512, 16, Wb1p, 256);
    packs(bW2, 256, 8,  Wb2p, 64);
    packs(tW0, 415, 13, Wt0p, 512);
    packs(tW1, 512, 16, Wt1p, 256);

    // ---- bottom MLP ----
    hipLaunchKernelGGL(cvt_dense, dim3(BATCH * 32 / 256), dim3(256), 0, stream, dense_x, A0);
    hipLaunchKernelGGL((gemm_wmma_relu<1>),  dim3(BATCH / 128, 8), dim3(128), 0, stream,
                       A0, 32, Wb0p, bb0, h1, 512);
    hipLaunchKernelGGL((gemm_wmma_relu<16>), dim3(BATCH / 128, 4), dim3(128), 0, stream,
                       h1, 512, Wb1p, bb1, h2, 256);
    // layer 3 writes directly into T row 0 of each sample (ldc = 1728)
    hipLaunchKernelGGL((gemm_wmma_relu<8>),  dim3(BATCH / 128, 1), dim3(128), 0, stream,
                       h2, 256, Wb2p, bb2, T, T_STRIDE);

    // ---- embedding gather into T rows 1..26 ----
    hipLaunchKernelGGL(gather_emb, dim3(NTAB * BATCH * 16 / 256), dim3(256), 0, stream,
                       lS_i, emb, T);

    // ---- pairwise-dot interaction -> R ----
    hipLaunchKernelGGL(interact_wmma, dim3(BATCH / 8), dim3(256), 0, stream, T, R);

    // ---- top MLP ----
    hipLaunchKernelGGL((gemm_wmma_relu<13>), dim3(BATCH / 128, 8), dim3(128), 0, stream,
                       R, R_STRIDE, Wt0p, tb0, th1, 512);
    hipLaunchKernelGGL((gemm_wmma_relu<16>), dim3(BATCH / 128, 4), dim3(128), 0, stream,
                       th1, 512, Wt1p, tb1, th2, 256);
    hipLaunchKernelGGL(final_sigmoid, dim3(BATCH / 256), dim3(256), 0, stream,
                       th2, tW2, tb2, out);
}
